// BinaryLutLayer_56367150793331
// MI455X (gfx1250) — 8-bit matrix operand order repaired and checked by reading, compiled, not run on hardware
//
#include <hip/hip_runtime.h>
#include <hip/hip_bf16.h>
#include <stdint.h>

#define NUM_BITS 14
#define NUM_OUT  16384
#define LUT_SIZE 16384

typedef __attribute__((ext_vector_type(8))) int v8i;

__global__ __launch_bounds__(128)
void binary_lut_wmma_kernel(const float* __restrict__ x,
                            const int8_t* __restrict__ luts_int,
                            float* __restrict__ out) {
    const int lane = (int)(threadIdx.x & 31u);
    const int wave = (int)(threadIdx.x >> 5u);
    const int tile = (int)blockIdx.x * 4 + wave;
    const int rowBase = tile * 16;

    const int mrow = lane & 15;
    const int half = lane >> 4;
    const int row  = rowBase + mrow;
    const int bitbase = half * 8;

    const float* __restrict__ xr = x + (size_t)row * NUM_BITS;
    unsigned a0 = 0u, a1 = 0u;
#pragma unroll
    for (int j = 0; j < 4; ++j) {
        unsigned bit = (unsigned)xr[bitbase + j];
        a0 |= bit << (8 * j);
    }
#pragma unroll
    for (int j = 0; j < 4; ++j) {
        int k  = bitbase + 4 + j;
        int kc = (k < NUM_BITS) ? k : (NUM_BITS - 1);
        unsigned bit = (unsigned)xr[kc];
        if (k >= NUM_BITS) bit = 0u;
        a1 |= bit << (8 * j);
    }
    const v8i A = (v8i){(int)a0, (int)a1, 0, 0, 0, 0, 0, 0};

    const int n  = lane & 15;
    const bool lowHalf = (half == 0);
    const int b0 = lowHalf ? ((n == 0) ? 0x08040201 : 0) : ((n == 1) ? 0x10080402 : 0);
    const int b1 = lowHalf ? ((n == 0) ? 0x00402010
                                       : ((n == 1) ? 0x01000000 : 0)) : ((n == 1) ? 0x00004020 : 0);
    const int b2 = 0;
    const int b3 = 0;
    const v8i B = (v8i){b0, b1, b2, b3, 0, 0, 0, 0};

    const v8i C = (v8i){0, 0, 0, 0, 0, 0, 0, 0};
    v8i D = __builtin_amdgcn_wmma_i32_16x16x64_iu8(false, A, false, B, C, false, false);

    int myaddr = 0;
#pragma unroll
    for (int r = 0; r < 8; ++r) {
        int lo0 = __shfl(D[r], 0, 32);
        int hi0 = __shfl(D[r], 1, 32);
        int lo1 = __shfl(D[r], 16, 32);
        int hi1 = __shfl(D[r], 17, 32);
        int addr0 = lo0 + (hi0 << 7);
        int addr1 = lo1 + (hi1 << 7);
        int sel = lowHalf ? addr0 : addr1;
        if ((lane & 7) == r) myaddr = sel;
    }

    if ((lane & 15) < 8) {
        int m    = (lane & 7) + half * 8;
        int orow = rowBase + m;
        int8_t v = luts_int[(size_t)orow * LUT_SIZE + (unsigned)myaddr];
        out[orow] = (float)v;
    }
}

extern "C" void kernel_launch(void* const* d_in, const int* in_sizes, int n_in,
                              void* d_out, int out_size, void* d_ws, size_t ws_size,
                              hipStream_t stream) {
    (void)in_sizes; (void)n_in; (void)d_ws; (void)ws_size; (void)out_size;
    const float*  x        = (const float*)d_in[0];
    const int8_t* luts_int = (const int8_t*)d_in[2];
    float* out = (float*)d_out;

    dim3 block(128);
    dim3 grid(NUM_OUT / 16 / 4);
    binary_lut_wmma_kernel<<<grid, block, 0, stream>>>(x, luts_int, out);
}
